// FeedForwardQuantum_65481071410799
// MI455X (gfx1250) — matrix operand order repaired and checked by reading, not compiled, not run on hardware
//
#include <hip/hip_runtime.h>

// ---------------------------------------------------------------------------
// Fused "quantum encoder -> FFN" for MI455X (gfx1250, wave32, WMMA).
//   q   = cos(x[:, :8]) * cos(sum(theta))           [M, 8]
//   h   = relu(q @ W1^T + b1)                       [M, 2048]  (f16, LDS only)
//   out = h @ W2^T + b2                             [M, 512]   (f32 accum)
//
// Block = 1024 threads (32 waves) = 128 rows x all 512 cols. Stage-1 on
// v_pk_fma_f16 with pre-packed W1/b1/q; stage-2 on v_wmma_f32_16x16x32_f16.
// hs/w2s double-buffered; next chunk produced after the WMMAs (XDL co-exec),
// one barrier per chunk. When scratch allows, W2 is pre-converted to f16 once
// and streamed into LDS with global_load_async_to_lds_b128 (ASYNCcnt).
// ---------------------------------------------------------------------------

typedef _Float16 v16h __attribute__((ext_vector_type(16)));
typedef float    v8f  __attribute__((ext_vector_type(8)));
typedef _Float16 h2v  __attribute__((ext_vector_type(2)));
typedef unsigned int u32;

constexpr int D_EMB = 512;
constexpr int FFN   = 2048;
constexpr int NQ    = 8;
constexpr int MTILE = 128;
constexpr int KC    = 32;
constexpr int LROW  = 40;        // halves per LDS row (80B = 5*16B, padded)
constexpr int NTHR  = 1024;      // 32 waves
constexpr int NCH   = FFN / KC;  // 64 chunks

__device__ __forceinline__ h2v bc2(u32 v) { return __builtin_bit_cast(h2v, v); }
__device__ __forceinline__ u32  pk(float a, float b) {
  return __builtin_bit_cast(u32, __builtin_amdgcn_cvt_pkrtz(a, b));
}

// One-time W2 f32 -> packed f16 (row-major [512][2048]) into scratch.
__global__ __launch_bounds__(256) void w2cvt(const float* __restrict__ W2,
                                             u32* __restrict__ dst) {
  int i = blockIdx.x * 256 + threadIdx.x;        // < D_EMB*FFN/2
  float2 v = ((const float2*)W2)[i];
  dst[i] = pk(v.x, v.y);
}

template <bool AW2>
__global__ __launch_bounds__(NTHR) void ffq_fused(
    const float* __restrict__ x, const float* __restrict__ theta,
    const float* __restrict__ W1, const float* __restrict__ b1,
    const float* __restrict__ W2, const float* __restrict__ b2,
    const _Float16* __restrict__ w2h,            // f16 W2 in scratch (AW2 only)
    float* __restrict__ out)
{
  // Persistent (read-only after init):
  __shared__ __align__(16) u32 qp[MTILE][NQ];      //  4 KB (q,q) packed
  __shared__ __align__(16) u32 w1p[FFN / 2][NQ];   // 32 KB W1 row-pair packed
  __shared__ __align__(16) u32 b1p[FFN / 2];       //  4 KB
  // Double-buffered WMMA operands:
  __shared__ __align__(16) _Float16 hs[2][MTILE][LROW];   // 20 KB
  __shared__ __align__(16) _Float16 w2s[2][D_EMB][LROW];  // 80 KB

  const int tid   = threadIdx.x;
  const int lane  = tid & 31;
  const int wv    = tid >> 5;
  const int strip = wv & 7;        // 16-row strip
  const int colq  = wv >> 3;       // 128-col quarter
  const int m0    = blockIdx.x * MTILE;

  float phi = 0.f;
#pragma unroll
  for (int j = 0; j < NQ; ++j) phi += theta[j];
  const float cphi = cosf(phi);

  // ---- one-time staging ----
  {
    int m = tid >> 3, j = tid & 7;
    float qf = cosf(x[(size_t)(m0 + m) * D_EMB + j]) * cphi;
    qp[m][j] = pk(qf, qf);
  }
  {
    int j = tid & 7, fpb = tid >> 3;
#pragma unroll
    for (int r = 0; r < 8; ++r) {
      int fp = fpb + (r << 7);
      float a = W1[(size_t)(2 * fp) * NQ + j];
      float b = W1[(size_t)(2 * fp + 1) * NQ + j];
      w1p[fp][j] = pk(a, b);
    }
  }
  {
    float2 bb = ((const float2*)b1)[tid];
    b1p[tid] = pk(bb.x, bb.y);
  }
  __syncthreads();

  const int kkp = (tid & 15) << 1;   // K-column pair in [0,32)
  const int rb  = tid >> 4;          // 0..63

  auto produce = [&](int k0, int b) {
    // hs: h = relu(q.W1 + b1), packed f16 pairs (v_pk_fma_f16 chain)
    const int fp = (k0 >> 1) + (tid & 15);
    const uint4* w1r4 = (const uint4*)&w1p[fp][0];
    uint4 wa = w1r4[0], wb = w1r4[1];
    h2v w0 = bc2(wa.x), w1h = bc2(wa.y), w2c = bc2(wa.z), w3 = bc2(wa.w);
    h2v w4 = bc2(wb.x), w5 = bc2(wb.y), w6 = bc2(wb.z), w7 = bc2(wb.w);
    h2v bv = bc2(b1p[fp]);
    const h2v zero = {(_Float16)0.f, (_Float16)0.f};
#pragma unroll
    for (int r = 0; r < 2; ++r) {
      int m = rb + (r << 6);
      const uint4* q4 = (const uint4*)&qp[m][0];
      uint4 qa = q4[0], qb = q4[1];
      h2v s = bv;
      s = __builtin_elementwise_fma(bc2(qa.x), w0, s);
      s = __builtin_elementwise_fma(bc2(qa.y), w1h, s);
      s = __builtin_elementwise_fma(bc2(qa.z), w2c, s);
      s = __builtin_elementwise_fma(bc2(qa.w), w3, s);
      s = __builtin_elementwise_fma(bc2(qb.x), w4, s);
      s = __builtin_elementwise_fma(bc2(qb.y), w5, s);
      s = __builtin_elementwise_fma(bc2(qb.z), w6, s);
      s = __builtin_elementwise_fma(bc2(qb.w), w7, s);
      s = __builtin_elementwise_max(s, zero);
      *reinterpret_cast<u32*>(&hs[b][m][kkp]) = __builtin_bit_cast(u32, s);
    }
    if constexpr (AW2) {
      // Async DMA: 512 rows x 64B of pre-converted f16 W2 -> LDS (padded rows)
#pragma unroll
      for (int r = 0; r < 2; ++r) {
        int t2 = tid + (r << 10);
        int n  = t2 >> 2;                 // 0..511
        int qq = t2 & 3;                  // 16B quarter of the 64B row
        u32 lofs = (u32)(uintptr_t)&w2s[b][n][qq * 8];
        u32 goff = (u32)((n * FFN + k0 + qq * 8) * 2);    // byte offset
        asm volatile("global_load_async_to_lds_b128 %0, %1, %2"
                     :: "v"(lofs), "v"(goff), "s"(w2h) : "memory");
      }
      asm volatile("s_wait_asynccnt 0x0" ::: "memory");
    } else {
      // Fallback: load f32 W2, convert+pack, store to LDS
#pragma unroll
      for (int r = 0; r < 8; ++r) {
        int n = rb + (r << 6);
        float2 v = *(const float2*)(W2 + (size_t)n * FFN + (k0 + kkp));
        *reinterpret_cast<u32*>(&w2s[b][n][kkp]) = pk(v.x, v.y);
      }
    }
  };

  produce(0, 0);
  __syncthreads();

  v8f acc[8] = {};   // wave tile: 16 rows x 128 cols

  for (int kc = 0; kc < NCH; ++kc) {
    const int cur = kc & 1;

    // A fragment (16x32 f16): lane m=l%16; elems = K 16h..16h+15
    const int h2   = lane >> 4;
    const int mrow = (strip << 4) + (lane & 15);
    v16h afrag;
#pragma unroll
    for (int j = 0; j < 8; ++j) afrag[j]     = hs[cur][mrow][h2 * 16 + j];
#pragma unroll
    for (int j = 0; j < 8; ++j) afrag[8 + j] = hs[cur][mrow][h2 * 16 + 8 + j];

    // 8 B fragments (32x16 f16): lane n=l%16; elems = K 16h..16h+15
#pragma unroll
    for (int c = 0; c < 8; ++c) {
      const int nrow = (colq << 7) + (c << 4) + (lane & 15);
      v16h bfrag;
#pragma unroll
      for (int j = 0; j < 16; ++j) bfrag[j] = w2s[cur][nrow][h2 * 16 + j];
      acc[c] = __builtin_amdgcn_wmma_f32_16x16x32_f16(
          false, afrag, false, bfrag, (short)0, acc[c], false, false);
    }

    if (kc + 1 < NCH) produce((kc + 1) * KC, cur ^ 1);  // uniform branch
    __syncthreads();
  }

  // Epilogue: D layout — N = lane%16, M = 8*(lane>=16) + vgpr index
  const int h2    = lane >> 4;
  const int mbase = m0 + (strip << 4) + (h2 << 3);
#pragma unroll
  for (int c = 0; c < 8; ++c) {
    int ncol = (colq << 7) + (c << 4) + (lane & 15);
    float bb = b2[ncol];
#pragma unroll
    for (int i = 0; i < 8; ++i)
      out[(size_t)(mbase + i) * D_EMB + ncol] = acc[c][i] + bb;
  }
}

extern "C" void kernel_launch(void* const* d_in, const int* in_sizes, int n_in,
                              void* d_out, int out_size, void* d_ws, size_t ws_size,
                              hipStream_t stream) {
  const float* x     = (const float*)d_in[0];
  const float* theta = (const float*)d_in[1];
  const float* W1    = (const float*)d_in[2];
  const float* b1    = (const float*)d_in[3];
  const float* W2    = (const float*)d_in[4];
  const float* b2    = (const float*)d_in[5];
  float* out = (float*)d_out;

  const int M = in_sizes[0] / D_EMB;                      // 65536 rows
  const size_t need = (size_t)D_EMB * FFN * sizeof(unsigned short);  // 2 MB
  dim3 grid(M / MTILE);                                   // 512 blocks

  if (ws_size >= need) {
    w2cvt<<<dim3((D_EMB * FFN / 2) / 256), 256, 0, stream>>>(W2, (u32*)d_ws);
    ffq_fused<true><<<grid, NTHR, 0, stream>>>(
        x, theta, W1, b1, W2, b2, (const _Float16*)d_ws, out);
  } else {
    ffq_fused<false><<<grid, NTHR, 0, stream>>>(
        x, theta, W1, b1, W2, b2, nullptr, out);
  }
}
